// TransformerRoPE_2516850835971
// MI455X (gfx1250) — matrix operand order repaired and checked by reading, not compiled, not run on hardware
//
#include <hip/hip_runtime.h>
#include <hip/hip_bf16.h>

// ---------------------------------------------------------------------------
// TransformerRoPE for MI455X (gfx1250): bf16 WMMA, f32 accumulate,
// async global->LDS staging (ASYNCcnt) for bf16-sourced tiles.
// B=2, L=2048, D=256, H=8, dh=32, 8 layers alternating self/cross attention.
// ---------------------------------------------------------------------------

#define BB 2
#define LL 2048
#define DD 256
#define NH 8
#define DH 32
#define NLAY 8
#define MROWS (BB * LL)        // 4096 rows for the projection GEMMs

typedef __attribute__((ext_vector_type(16))) __bf16 v16bf;
typedef __attribute__((ext_vector_type(8)))  float  v8f;

// 16-bit A-matrix (16x32, MxK) element->K map  (ISA 7.12.2)
__device__ __forceinline__ int a_kidx(int e, int half) {
    return e + (half << 3) + ((e >> 3) << 3);
}
// 16-bit B-matrix (32x16, KxN) element->K map (same order as A)
__device__ __forceinline__ int b_kidx(int e, int half) {
    return e + (half << 3) + ((e >> 3) << 3);
}

// --- gfx1250 async global->LDS copies (bypass VGPRs, tracked by ASYNCcnt) ---
__device__ __forceinline__ void async_copy_b64(unsigned lds_addr, const void* gptr) {
    asm volatile("global_load_async_to_lds_b64 %0, %1, off"
                 :: "v"(lds_addr), "v"(gptr) : "memory");
}
__device__ __forceinline__ void async_copy_b128(unsigned lds_addr, const void* gptr) {
    asm volatile("global_load_async_to_lds_b128 %0, %1, off"
                 :: "v"(lds_addr), "v"(gptr) : "memory");
}
__device__ __forceinline__ void wait_async0() {
#if __has_builtin(__builtin_amdgcn_s_wait_asynccnt)
    __builtin_amdgcn_s_wait_asynccnt(0);
#else
    asm volatile("s_wait_asynccnt 0x0" ::: "memory");
#endif
}
// low 32 bits of a flat pointer to __shared__ == LDS byte address
__device__ __forceinline__ unsigned lds_off(const void* p) {
    return (unsigned)(size_t)p;
}

// ---------------------------------------------------------------------------
// GEMM: C(bf16, M x 256) = A(f32, M x 256) @ W^T      (W f32, 256 x 256)
// Block: 256 threads = 8 waves, block tile 64(M) x 64(N), wave tile 16x32.
// fp32 source must be converted, so staging stays on the VALU path; prefetch
// the next K-chunk to cover latency.
// ---------------------------------------------------------------------------
__global__ __launch_bounds__(256)
void gemm_f32_wT_bf16(const float* __restrict__ A, const float* __restrict__ W,
                      __bf16* __restrict__ C) {
    __shared__ __bf16 As[64][40];
    __shared__ __bf16 Ws[64][40];
    const int tid  = threadIdx.x;
    const int m0   = blockIdx.x * 64;
    const int n0   = blockIdx.y * 64;
    const int wave = tid >> 5, lane = tid & 31;
    const int half = lane >> 4, mr = lane & 15;
    const int wm   = wave & 3,  wn = wave >> 2;

    v8f acc0 = {}, acc1 = {};
    for (int kc = 0; kc < DD; kc += 32) {
#pragma unroll
        for (int j = 0; j < 8; ++j) {          // 64x32 A chunk, f32 -> bf16
            int idx = tid + j * 256;
            int r = idx >> 5, c = idx & 31;
            As[r][c] = (__bf16)A[(size_t)(m0 + r) * DD + kc + c];
            Ws[r][c] = (__bf16)W[(size_t)(n0 + r) * DD + kc + c];
        }
        if (kc + 32 < DD) {                    // prefetch next chunk
            __builtin_prefetch(&A[(size_t)(m0 + (tid >> 5)) * DD + kc + 32], 0, 3);
            __builtin_prefetch(&W[(size_t)(n0 + (tid >> 5)) * DD + kc + 32], 0, 3);
        }
        __syncthreads();
        v16bf a, b0, b1;
#pragma unroll
        for (int e = 0; e < 16; ++e) {
            a[e]  = As[wm * 16 + mr][a_kidx(e, half)];
            int k = b_kidx(e, half);
            b0[e] = Ws[wn * 32 + mr][k];
            b1[e] = Ws[wn * 32 + 16 + mr][k];
        }
        acc0 = __builtin_amdgcn_wmma_f32_16x16x32_bf16(false, a, false, b0,
                                                       (short)0, acc0, false, false);
        acc1 = __builtin_amdgcn_wmma_f32_16x16x32_bf16(false, a, false, b1,
                                                       (short)0, acc1, false, false);
        __syncthreads();
    }
#pragma unroll
    for (int r = 0; r < 8; ++r) {
        size_t row = (size_t)(m0 + wm * 16 + r + half * 8) * DD;
        int col = n0 + wn * 32 + mr;
        C[row + col]      = (__bf16)acc0[r];
        C[row + col + 16] = (__bf16)acc1[r];
    }
}

// ---------------------------------------------------------------------------
// GEMM + residual: F(f32, in/out) = F + A(bf16) @ W^T
// A is already bf16 in global: stage it with async global->LDS B128 copies.
// ---------------------------------------------------------------------------
__global__ __launch_bounds__(256)
void gemm_bf16_wT_residual(const __bf16* __restrict__ A, const float* __restrict__ W,
                           float* __restrict__ F) {
    __shared__ __bf16 As[64][40];
    __shared__ __bf16 Ws[64][40];
    const int tid  = threadIdx.x;
    const int m0   = blockIdx.x * 64;
    const int n0   = blockIdx.y * 64;
    const int wave = tid >> 5, lane = tid & 31;
    const int half = lane >> 4, mr = lane & 15;
    const int wm   = wave & 3,  wn = wave >> 2;

    v8f acc0 = {}, acc1 = {};
    for (int kc = 0; kc < DD; kc += 32) {
        {   // async-stage 64x32 bf16 A chunk: 16B per thread, no VGPR round trip
            int r = tid >> 2, c = (tid & 3) * 8;
            async_copy_b128(lds_off(&As[r][c]),
                            &A[(size_t)(m0 + r) * DD + kc + c]);
        }
#pragma unroll
        for (int j = 0; j < 8; ++j) {          // W chunk needs f32 -> bf16
            int idx = tid + j * 256;
            int r = idx >> 5, c = idx & 31;
            Ws[r][c] = (__bf16)W[(size_t)(n0 + r) * DD + kc + c];
        }
        wait_async0();
        __syncthreads();
        v16bf a, b0, b1;
#pragma unroll
        for (int e = 0; e < 16; ++e) {
            a[e]  = As[wm * 16 + mr][a_kidx(e, half)];
            int k = b_kidx(e, half);
            b0[e] = Ws[wn * 32 + mr][k];
            b1[e] = Ws[wn * 32 + 16 + mr][k];
        }
        acc0 = __builtin_amdgcn_wmma_f32_16x16x32_bf16(false, a, false, b0,
                                                       (short)0, acc0, false, false);
        acc1 = __builtin_amdgcn_wmma_f32_16x16x32_bf16(false, a, false, b1,
                                                       (short)0, acc1, false, false);
        __syncthreads();
    }
#pragma unroll
    for (int r = 0; r < 8; ++r) {
        size_t row = (size_t)(m0 + wm * 16 + r + half * 8) * DD;
        int col = n0 + wn * 32 + mr;
        F[row + col]      = F[row + col]      + acc0[r];
        F[row + col + 16] = F[row + col + 16] + acc1[r];
    }
}

// ---------------------------------------------------------------------------
// RoPE (in place, full-D rotation before head split, matching the reference).
// ---------------------------------------------------------------------------
__global__ __launch_bounds__(256)
void rope_kernel(__bf16* __restrict__ X) {
    int idx = blockIdx.x * 256 + threadIdx.x;       // B*L*128 threads
    int j  = idx & 127;
    int bl = idx >> 7;                              // b*L + t
    int t  = bl & (LL - 1);
    float inv_freq = __expf(-((float)(2 * j) * (1.0f / 256.0f)) * 9.210340371976184f);
    float f = (float)t * inv_freq;
    float c = __cosf(f), s = __sinf(f);
    __bf16* row = X + (size_t)bl * DD;
    float x1 = (float)row[j];
    float x2 = (float)row[j + 128];
    row[j]       = (__bf16)(x1 * c - x2 * s);
    row[j + 128] = (__bf16)(x2 * c + x1 * s);
}

// ---------------------------------------------------------------------------
// Flash attention: grid (B*H, L/128), 256 threads = 8 waves, one 16-row Q
// tile per wave. K/V chunks staged with async global->LDS copies (bf16, no
// conversion needed). Online softmax via intra-halfwave shfl_xor reductions.
// ---------------------------------------------------------------------------
__global__ __launch_bounds__(256)
void attn_fa_kernel(const __bf16* __restrict__ Q, const __bf16* __restrict__ Kt,
                    const __bf16* __restrict__ V, __bf16* __restrict__ O) {
    __shared__ __bf16 Ks[32][40];
    __shared__ __bf16 Vs[32][40];
    __shared__ __bf16 Ps[8][16][40];

    const int tid  = threadIdx.x;
    const int wave = tid >> 5, lane = tid & 31;
    const int half = lane >> 4, mr = lane & 15;
    const int b = blockIdx.x >> 3;
    const int h = blockIdx.x & 7;
    const int q0 = blockIdx.y * 128 + wave * 16;
    const float scale = 0.17677669529663687f;       // 32^-0.5

    const size_t qbase = ((size_t)b * LL + q0) * DD + h * DH;

    v16bf qa;
#pragma unroll
    for (int e = 0; e < 16; ++e)
        qa[e] = (__bf16)((float)Q[qbase + (size_t)mr * DD + a_kidx(e, half)] * scale);

    v8f o0 = {}, o1 = {};
    float runM[8], runL[8];
#pragma unroll
    for (int r = 0; r < 8; ++r) { runM[r] = -3.0e38f; runL[r] = 0.0f; }

    // staging assignment: 8 threads per kv row, 4 bf16 (8B) each
    const int sr = tid >> 3, sc = (tid & 7) * 4;
    const unsigned ldsK = lds_off(&Ks[sr][sc]);
    const unsigned ldsV = lds_off(&Vs[sr][sc]);

    for (int kv0 = 0; kv0 < LL; kv0 += 32) {
        size_t g = ((size_t)b * LL + kv0 + sr) * DD + h * DH + sc;
        async_copy_b64(ldsK, &Kt[g]);
        async_copy_b64(ldsV, &V[g]);
        wait_async0();
        __syncthreads();

        // scores: S = Q @ K^T   (two 16x16 tiles covering 32 kv positions)
        v16bf kb0, kb1;
#pragma unroll
        for (int e = 0; e < 16; ++e) {
            int k = b_kidx(e, half);
            kb0[e] = Ks[mr][k];
            kb1[e] = Ks[16 + mr][k];
        }
        v8f s0 = {}, s1 = {};
        s0 = __builtin_amdgcn_wmma_f32_16x16x32_bf16(false, qa, false, kb0,
                                                     (short)0, s0, false, false);
        s1 = __builtin_amdgcn_wmma_f32_16x16x32_bf16(false, qa, false, kb1,
                                                     (short)0, s1, false, false);

        // online softmax; row m = r + 8*half lives in one 16-lane half
#pragma unroll
        for (int r = 0; r < 8; ++r) {
            float tm = fmaxf(s0[r], s1[r]);
#pragma unroll
            for (int m = 1; m < 16; m <<= 1) tm = fmaxf(tm, __shfl_xor(tm, m, 32));
            float nm = fmaxf(runM[r], tm);
            float corr = __expf(runM[r] - nm);
            runM[r] = nm;
            runL[r] *= corr;
            o0[r] *= corr;
            o1[r] *= corr;
            float p0 = __expf(s0[r] - nm);
            float p1 = __expf(s1[r] - nm);
            float ps = p0 + p1;
#pragma unroll
            for (int m = 1; m < 16; m <<= 1) ps += __shfl_xor(ps, m, 32);
            runL[r] += ps;
            int mrow = r + half * 8;
            Ps[wave][mrow][mr]      = (__bf16)p0;
            Ps[wave][mrow][16 + mr] = (__bf16)p1;
        }
        __syncthreads();

        // O += P @ V   (two 16x16 output tiles covering dh=32)
        v16bf pa, vb0, vb1;
#pragma unroll
        for (int e = 0; e < 16; ++e) {
            pa[e] = Ps[wave][mr][a_kidx(e, half)];
            int k = b_kidx(e, half);
            vb0[e] = Vs[k][mr];
            vb1[e] = Vs[k][16 + mr];
        }
        o0 = __builtin_amdgcn_wmma_f32_16x16x32_bf16(false, pa, false, vb0,
                                                     (short)0, o0, false, false);
        o1 = __builtin_amdgcn_wmma_f32_16x16x32_bf16(false, pa, false, vb1,
                                                     (short)0, o1, false, false);
        __syncthreads();
    }

#pragma unroll
    for (int r = 0; r < 8; ++r) {
        float inv = 1.0f / runL[r];
        size_t orow = qbase + (size_t)(r + half * 8) * DD;
        O[orow + mr]      = (__bf16)(o0[r] * inv);
        O[orow + 16 + mr] = (__bf16)(o1[r] * inv);
    }
}

// ---------------------------------------------------------------------------
// Host side
// ---------------------------------------------------------------------------
static void run_layer(float* x, const float* src, const float* wq, const float* wk,
                      const float* wv, const float* wo, bool use_rope,
                      __bf16* Qb, __bf16* Kb, __bf16* Vb, __bf16* Ab,
                      hipStream_t stream) {
    dim3 gg(MROWS / 64, DD / 64);   // (64, 4)
    gemm_f32_wT_bf16<<<gg, 256, 0, stream>>>(x,   wq, Qb);
    gemm_f32_wT_bf16<<<gg, 256, 0, stream>>>(src, wk, Kb);
    gemm_f32_wT_bf16<<<gg, 256, 0, stream>>>(src, wv, Vb);
    if (use_rope) {
        rope_kernel<<<(MROWS * 128) / 256, 256, 0, stream>>>(Qb);
        rope_kernel<<<(MROWS * 128) / 256, 256, 0, stream>>>(Kb);
    }
    attn_fa_kernel<<<dim3(BB * NH, LL / 128), 256, 0, stream>>>(Qb, Kb, Vb, Ab);
    gemm_bf16_wT_residual<<<gg, 256, 0, stream>>>(Ab, wo, x);
}

extern "C" void kernel_launch(void* const* d_in, const int* in_sizes, int n_in,
                              void* d_out, int out_size, void* d_ws, size_t ws_size,
                              hipStream_t stream) {
    const float* f0_in = (const float*)d_in[0];
    const float* f1_in = (const float*)d_in[1];
    const float* Wq    = (const float*)d_in[2];
    const float* Wk    = (const float*)d_in[3];
    const float* Wv    = (const float*)d_in[4];
    const float* Wo    = (const float*)d_in[5];

    const size_t FELEMS = (size_t)BB * LL * DD;
    const size_t FBYTES = FELEMS * sizeof(float);
    const size_t HBYTES = FELEMS * sizeof(__bf16);

    char* ws = (char*)d_ws;
    float*  f0 = (float*)ws;  ws += FBYTES;
    float*  f1 = (float*)ws;  ws += FBYTES;
    __bf16* Qb = (__bf16*)ws; ws += HBYTES;
    __bf16* Kb = (__bf16*)ws; ws += HBYTES;
    __bf16* Vb = (__bf16*)ws; ws += HBYTES;
    __bf16* Ab = (__bf16*)ws; ws += HBYTES;

    hipMemcpyAsync(f0, f0_in, FBYTES, hipMemcpyDeviceToDevice, stream);
    hipMemcpyAsync(f1, f1_in, FBYTES, hipMemcpyDeviceToDevice, stream);

    for (int i = 0; i < NLAY; ++i) {
        const float* wq = Wq + (size_t)i * DD * DD;
        const float* wk = Wk + (size_t)i * DD * DD;
        const float* wv = Wv + (size_t)i * DD * DD;
        const float* wo = Wo + (size_t)i * DD * DD;
        if ((i & 1) == 0) {                 // self-attention (with RoPE)
            run_layer(f0, f0, wq, wk, wv, wo, true,  Qb, Kb, Vb, Ab, stream);
            run_layer(f1, f1, wq, wk, wv, wo, true,  Qb, Kb, Vb, Ab, stream);
        } else {                            // cross-attention (f1 sees updated f0)
            run_layer(f0, f1, wq, wk, wv, wo, false, Qb, Kb, Vb, Ab, stream);
            run_layer(f1, f0, wq, wk, wv, wo, false, Qb, Kb, Vb, Ab, stream);
        }
    }

    hipMemcpyAsync(d_out, f0, FBYTES, hipMemcpyDeviceToDevice, stream);
    hipMemcpyAsync((char*)d_out + FBYTES, f1, FBYTES, hipMemcpyDeviceToDevice, stream);
}
